// BasicInteractionNetworkModule_76940044141039
// MI455X (gfx1250) — matrix operand order repaired and checked by reading, not compiled, not run on hardware
//
#include <hip/hip_runtime.h>

typedef __bf16 bf16;
typedef __attribute__((ext_vector_type(8)))  bf16  bf16x8;
typedef __attribute__((ext_vector_type(16))) bf16  bf16x16;
typedef __attribute__((ext_vector_type(8)))  float f32x8;
typedef __attribute__((ext_vector_type(4)))  int   intx4;

#define TILE_M 128
#define TILE_N 128
#define K_STEP 64
#define LDS_LD 72   // LDS row stride in bf16 elements: 64 data + 8 pad (bank spread)

#if __has_builtin(__builtin_amdgcn_global_load_async_to_lds_b128)
#define HAVE_ASYNC_LDS 1
#endif

#if __has_builtin(__builtin_amdgcn_s_wait_asynccnt)
#define WAIT_ASYNC() __builtin_amdgcn_s_wait_asynccnt(0)
#else
#define WAIT_ASYNC() asm volatile("s_wait_asynccnt 0" ::: "memory")
#endif

#ifdef HAVE_ASYNC_LDS
// async DMA of 16B global -> LDS; imm offset applies to both addresses (ISA §8.4.4)
// signature (probe-derived): (int4 AS1*, int4 AS3*, imm offset, imm cpol)
#define ASYNC_CP16(gp, lp, off)                                             \
  __builtin_amdgcn_global_load_async_to_lds_b128(                           \
      (__attribute__((address_space(1))) intx4*)(gp),                       \
      (__attribute__((address_space(3))) intx4*)(lp), (off), 0)
#endif

// ---------------------------------------------------------------------------
// Generic bf16 GEMM:  C[m][n] = sum_k A[m][k] * BT[n][k]  (+bias, optional ReLU)
// A  : row-major [M x K] bf16, leading dim lda
// BT : row-major [N x K] bf16 (B pre-transposed), leading dim ldbt
// C  : row-major, leading dim ldc; f32 or bf16 output
// Batched via blockIdx.z with element strides sA/sBT/sC.
// M, N multiples of 128; K multiple of 64 (guaranteed by caller).
// ---------------------------------------------------------------------------
template<bool RELU, bool OUT_F32>
__global__ __launch_bounds__(256) void gemm_bt_kernel(
    const bf16* __restrict__ A, const bf16* __restrict__ BT,
    const float* __restrict__ bias, void* __restrict__ Cout,
    int K, int lda, int ldbt, int ldc,
    long long sA, long long sBT, long long sC)
{
  __shared__ __align__(16) bf16 As[2][TILE_M * LDS_LD];
  __shared__ __align__(16) bf16 Bs[2][TILE_N * LDS_LD];

  const int b = blockIdx.z;
  A  += (long long)b * sA;
  BT += (long long)b * sBT;

  const int m0 = blockIdx.x * TILE_M;
  const int n0 = blockIdx.y * TILE_N;

  const int t     = threadIdx.x;
  const int lane  = t & 31;
  const int wave  = t >> 5;
  const int wm    = (wave & 3) * 32;   // wave's M offset within tile (4 waves over M)
  const int wn    = (wave >> 2) * 64;  // wave's N offset within tile (2 waves over N)
  const int lmod  = lane & 15;
  const int lhalf = lane >> 4;

  // staging: 2 threads per 128B tile row, each owns a contiguous 64B chunk
  const int sr = t >> 1;               // tile row 0..127
  const int sh = (t & 1) * 32;         // element offset of this thread's chunk

  const bf16* gA = A  + (long long)(m0 + sr) * lda  + sh;
  const bf16* gB = BT + (long long)(n0 + sr) * ldbt + sh;
  const int lofs = sr * LDS_LD + sh;   // LDS element offset of this thread's chunk

  f32x8 acc[2][4];
#pragma unroll
  for (int mi = 0; mi < 2; ++mi)
#pragma unroll
    for (int ni = 0; ni < 4; ++ni)
#pragma unroll
      for (int r = 0; r < 8; ++r) acc[mi][ni][r] = 0.0f;

  // two 32-wide WMMA k-slices over the 64-wide staged buffer
  auto compute = [&](int buf) {
#pragma unroll
    for (int ks = 0; ks < K_STEP; ks += 32) {
      bf16x16 af[2];
#pragma unroll
      for (int mi = 0; mi < 2; ++mi) {
        const bf16* p = &As[buf][(wm + mi * 16 + lmod) * LDS_LD + ks + lhalf * 8];
        bf16x8 lo = *(const bf16x8*)p;
        bf16x8 hi = *(const bf16x8*)(p + 16);
        af[mi] = __builtin_shufflevector(lo, hi, 0,1,2,3,4,5,6,7,8,9,10,11,12,13,14,15);
      }
      bf16x16 bfr[4];
#pragma unroll
      for (int ni = 0; ni < 4; ++ni) {
        const bf16* p = &Bs[buf][(wn + ni * 16 + lmod) * LDS_LD + ks + lhalf * 8];
        bf16x8 lo = *(const bf16x8*)p;
        bf16x8 hi = *(const bf16x8*)(p + 16);
        bfr[ni] = __builtin_shufflevector(lo, hi, 0,1,2,3,4,5,6,7,8,9,10,11,12,13,14,15);
      }
#pragma unroll
      for (int mi = 0; mi < 2; ++mi)
#pragma unroll
        for (int ni = 0; ni < 4; ++ni)
          acc[mi][ni] = __builtin_amdgcn_wmma_f32_16x16x32_bf16(
              false, af[mi], false, bfr[ni], (short)0, acc[mi][ni], false, false);
    }
  };

#ifdef HAVE_ASYNC_LDS
  // ---- double-buffered async (ASYNCcnt) pipeline: 1 barrier / K-slab ----
  auto issue = [&](int buf, int kk) {
    const char* ga = (const char*)(gA + kk);
    const char* gb = (const char*)(gB + kk);
    bf16* la = &As[buf][lofs];
    bf16* lb = &Bs[buf][lofs];
    ASYNC_CP16(ga, la, 0);  ASYNC_CP16(ga, la, 16);
    ASYNC_CP16(ga, la, 32); ASYNC_CP16(ga, la, 48);
    ASYNC_CP16(gb, lb, 0);  ASYNC_CP16(gb, lb, 16);
    ASYNC_CP16(gb, lb, 32); ASYNC_CP16(gb, lb, 48);
  };
  int buf = 0;
  issue(0, 0);
  for (int kk = 0; kk < K; kk += K_STEP) {
    WAIT_ASYNC();        // own slab landed in LDS
    __syncthreads();     // everyone's slab landed; prev slab's reads all retired
    if (kk + K_STEP < K) issue(buf ^ 1, kk + K_STEP);
    if (kk + 2 * K_STEP < K) {  // warm L2/WGP$ for the slab after next
      __builtin_prefetch((const char*)(gA + kk) + 4 * K_STEP, 0, 3);
      __builtin_prefetch((const char*)(gB + kk) + 4 * K_STEP, 0, 3);
    }
    compute(buf);
    buf ^= 1;
  }
#else
  // ---- fallback: synchronous staging through VGPRs ----
  for (int kk = 0; kk < K; kk += K_STEP) {
    const uint4* ga = (const uint4*)(gA + kk);
    const uint4* gb = (const uint4*)(gB + kk);
    uint4 a0 = ga[0], a1 = ga[1], a2 = ga[2], a3 = ga[3];
    uint4 b0 = gb[0], b1 = gb[1], b2 = gb[2], b3 = gb[3];
    if (kk + K_STEP < K) {
      __builtin_prefetch((const char*)ga + 2 * K_STEP, 0, 3);
      __builtin_prefetch((const char*)gb + 2 * K_STEP, 0, 3);
    }
    __syncthreads();
    {
      uint4* da = (uint4*)&As[0][lofs];
      da[0] = a0; da[1] = a1; da[2] = a2; da[3] = a3;
      uint4* db = (uint4*)&Bs[0][lofs];
      db[0] = b0; db[1] = b1; db[2] = b2; db[3] = b3;
    }
    __syncthreads();
    compute(0);
  }
#endif

  // C/D layout: VGPR r -> lanes 0-15: (M=r, N=lane); lanes 16-31: (M=8+r, N=lane-16)
#pragma unroll
  for (int mi = 0; mi < 2; ++mi) {
    const int rbase = m0 + wm + mi * 16 + lhalf * 8;
#pragma unroll
    for (int ni = 0; ni < 4; ++ni) {
      const int col = n0 + wn + ni * 16 + lmod;
      const float bv = bias ? bias[col] : 0.0f;
#pragma unroll
      for (int r = 0; r < 8; ++r) {
        float v = acc[mi][ni][r] + bv;
        if (RELU) v = fmaxf(v, 0.0f);
        const long long idx = (long long)b * sC + (long long)(rbase + r) * ldc + col;
        if (OUT_F32) ((float*)Cout)[idx] = v;
        else         ((bf16*)Cout)[idx]  = (bf16)v;
      }
    }
  }
}

// ---------------------------------------------------------------------------
// Tiled transpose + cast to bf16:  src [R x C] (T) -> dst [C x R], dst ld = ldd
// R, C multiples of 32. Batched via blockIdx.z.
// ---------------------------------------------------------------------------
template<typename T>
__global__ __launch_bounds__(256) void transpose_cast_kernel(
    const T* __restrict__ src, bf16* __restrict__ dst,
    int R, int C, int ldd, long long sSrc, long long sDst)
{
  __shared__ bf16 tile[32][33];
  const int b = blockIdx.z;
  src += (long long)b * sSrc;
  dst += (long long)b * sDst;
  const int c0 = blockIdx.x * 32;
  const int r0 = blockIdx.y * 32;
  for (int i = threadIdx.y; i < 32; i += 8)
    tile[i][threadIdx.x] = (bf16)src[(long long)(r0 + i) * C + (c0 + threadIdx.x)];
  __syncthreads();
  for (int i = threadIdx.y; i < 32; i += 8)
    dst[(long long)(c0 + i) * ldd + (r0 + threadIdx.x)] = tile[threadIdx.x][i];
}

// ---------------------------------------------------------------------------
// Strided cast-copy f32 -> bf16 (splices tensors into concat buffers)
// ---------------------------------------------------------------------------
__global__ __launch_bounds__(256) void copy_cast_kernel(
    const float* __restrict__ src, bf16* __restrict__ dst,
    long long total, int cols, int lds, int ldd)
{
  long long i = (long long)blockIdx.x * 256 + threadIdx.x;
  if (i >= total) return;
  long long r = i / cols;
  int c = (int)(i - r * cols);
  dst[r * (long long)ldd + c] = (bf16)src[r * (long long)lds + c];
}

// zero-fill a column stripe [all rows] x [c0, c0+cols) of a bf16 matrix
__global__ __launch_bounds__(256) void zero_pad_kernel(
    bf16* __restrict__ dst, long long total, int cols, int ld, int c0)
{
  long long i = (long long)blockIdx.x * 256 + threadIdx.x;
  if (i >= total) return;
  long long r = i / cols;
  int c = (int)(i - r * cols);
  dst[r * (long long)ld + c0 + c] = (bf16)0.0f;
}

extern "C" void kernel_launch(void* const* d_in, const int* in_sizes, int n_in,
                              void* d_out, int out_size, void* d_ws, size_t ws_size,
                              hipStream_t stream)
{
  (void)in_sizes; (void)n_in; (void)out_size; (void)ws_size;

  const float* objects  = (const float*)d_in[0];
  const float* send_rel = (const float*)d_in[1];
  const float* recv_rel = (const float*)d_in[2];
  const float* rel_info = (const float*)d_in[3];
  const float* ext_eff  = (const float*)d_in[4];
  const float* rw1 = (const float*)d_in[5];  const float* rb1 = (const float*)d_in[6];
  const float* rw2 = (const float*)d_in[7];  const float* rb2 = (const float*)d_in[8];
  const float* rw3 = (const float*)d_in[9];  const float* rb3 = (const float*)d_in[10];
  const float* rw4 = (const float*)d_in[11]; const float* rb4 = (const float*)d_in[12];
  const float* ow1 = (const float*)d_in[13]; const float* ob1 = (const float*)d_in[14];
  const float* ow2 = (const float*)d_in[15]; const float* ob2 = (const float*)d_in[16];
  const float* ow3 = (const float*)d_in[17]; const float* ob3 = (const float*)d_in[18];
  float* out = (float*)d_out;

  constexpr int Bb = 16, NOBJ = 128, NREL = 16256;
  constexpr int OBJD = 128, RELD = 64, EFFD = 128, EXTD = 32, OUTD = 128, HID = 256;
  constexpr long long MREL = (long long)Bb * NREL;  // 260096
  constexpr long long MOBJ = (long long)Bb * NOBJ;  // 2048
  constexpr int RELIN  = 2 * OBJD + RELD;           // 320
  constexpr int OBJIN  = OBJD + EXTD + EFFD;        // 288
  constexpr int OBJINP = 320;                       // padded to K%64==0

  char* ws = (char*)d_ws;
  size_t off = 0;
  auto take = [&](size_t bytes) -> void* {
    void* p = (void*)(ws + off);
    off += (bytes + 255) & ~(size_t)255;
    return p;
  };
  bf16*  ST   = (bf16*)take((size_t)Bb * NREL * NOBJ * 2);   // S^T  [b][r][o]
  bf16*  RT   = (bf16*)take((size_t)Bb * NREL * NOBJ * 2);   // R^T  [b][r][o]
  bf16*  Rb   = (bf16*)take((size_t)Bb * NOBJ * NREL * 2);   // R    [b][o][r]
  bf16*  OBJT = (bf16*)take((size_t)Bb * OBJD * NOBJ * 2);   // O^T  [b][d][o]
  bf16*  rwT1 = (bf16*)take((size_t)HID * RELIN * 2);
  bf16*  rwT2 = (bf16*)take((size_t)HID * HID * 2);
  bf16*  rwT3 = (bf16*)take((size_t)HID * HID * 2);
  bf16*  rwT4 = (bf16*)take((size_t)EFFD * HID * 2);
  bf16*  owT1 = (bf16*)take((size_t)HID * OBJINP * 2);       // padded [256][320]
  bf16*  owT2 = (bf16*)take((size_t)HID * HID * 2);
  bf16*  owT3 = (bf16*)take((size_t)OUTD * HID * 2);
  bf16*  relx = (bf16*)take((size_t)MREL * RELIN * 2);       // [senders|receivers|rel_info]
  bf16*  hA   = (bf16*)take((size_t)MREL * HID * 2);
  bf16*  hB   = (bf16*)take((size_t)MREL * HID * 2);
  float* effr = (float*)take((size_t)Bb * NOBJ * EFFD * 4);  // effect_receivers f32
  bf16*  objx = (bf16*)take((size_t)MOBJ * OBJINP * 2);      // padded [2048][320]
  bf16*  gA   = (bf16*)take((size_t)MOBJ * HID * 2);
  bf16*  gB   = (bf16*)take((size_t)MOBJ * HID * 2);
  bf16*  effects = relx;  // alias: rel_x fully consumed by MLP layer 1
  bf16*  effT    = hB;    // alias: hB free after MLP layer 3

  dim3 tb(32, 8);

  // --- convert / transpose inputs & weights to bf16 -------------------------
  transpose_cast_kernel<float><<<dim3(NREL/32, NOBJ/32, Bb), tb, 0, stream>>>(
      send_rel, ST, NOBJ, NREL, NOBJ, (long long)NOBJ*NREL, (long long)NREL*NOBJ);
  transpose_cast_kernel<float><<<dim3(NREL/32, NOBJ/32, Bb), tb, 0, stream>>>(
      recv_rel, RT, NOBJ, NREL, NOBJ, (long long)NOBJ*NREL, (long long)NREL*NOBJ);
  transpose_cast_kernel<float><<<dim3(OBJD/32, NOBJ/32, Bb), tb, 0, stream>>>(
      objects, OBJT, NOBJ, OBJD, NOBJ, (long long)NOBJ*OBJD, (long long)OBJD*NOBJ);
  transpose_cast_kernel<float><<<dim3(HID/32,  RELIN/32, 1), tb, 0, stream>>>(rw1, rwT1, RELIN, HID, RELIN,  0, 0);
  transpose_cast_kernel<float><<<dim3(HID/32,  HID/32,   1), tb, 0, stream>>>(rw2, rwT2, HID,   HID, HID,    0, 0);
  transpose_cast_kernel<float><<<dim3(HID/32,  HID/32,   1), tb, 0, stream>>>(rw3, rwT3, HID,   HID, HID,    0, 0);
  transpose_cast_kernel<float><<<dim3(EFFD/32, HID/32,   1), tb, 0, stream>>>(rw4, rwT4, HID,   EFFD, HID,   0, 0);
  transpose_cast_kernel<float><<<dim3(HID/32,  OBJIN/32, 1), tb, 0, stream>>>(ow1, owT1, OBJIN, HID, OBJINP, 0, 0);
  transpose_cast_kernel<float><<<dim3(HID/32,  HID/32,   1), tb, 0, stream>>>(ow2, owT2, HID,   HID, HID,    0, 0);
  transpose_cast_kernel<float><<<dim3(OUTD/32, HID/32,   1), tb, 0, stream>>>(ow3, owT3, HID,   OUTD, HID,   0, 0);

  // zero the K-pad stripes (poisoned workspace otherwise)
  zero_pad_kernel<<<(unsigned)((HID*(OBJINP-OBJIN) + 255)/256), 256, 0, stream>>>(
      owT1, (long long)HID*(OBJINP-OBJIN), OBJINP-OBJIN, OBJINP, OBJIN);
  zero_pad_kernel<<<(unsigned)((MOBJ*(OBJINP-OBJIN) + 255)/256), 256, 0, stream>>>(
      objx, MOBJ*(OBJINP-OBJIN), OBJINP-OBJIN, OBJINP, OBJIN);

  // relation_info -> rel_x[:, 256:320]
  {
    long long total = MREL * RELD;
    copy_cast_kernel<<<(unsigned)((total + 255) / 256), 256, 0, stream>>>(
        rel_info, relx + 2 * OBJD, total, RELD, RELD, RELIN);
  }
  // receiver_relations f32 -> Rb bf16 (aggregation GEMM A-operand)
  {
    long long total = (long long)Bb * NOBJ * NREL;
    copy_cast_kernel<<<(unsigned)((total + 255) / 256), 256, 0, stream>>>(
        recv_rel, Rb, total, NREL, NREL, NREL);
  }

  // --- senders / receivers: (R^T @ O) into rel_x columns --------------------
  gemm_bt_kernel<false,false><<<dim3(NREL/TILE_M, OBJD/TILE_N, Bb), 256, 0, stream>>>(
      ST, OBJT, nullptr, (void*)relx, NOBJ, NOBJ, NOBJ, RELIN,
      (long long)NREL*NOBJ, (long long)OBJD*NOBJ, (long long)NREL*RELIN);
  gemm_bt_kernel<false,false><<<dim3(NREL/TILE_M, OBJD/TILE_N, Bb), 256, 0, stream>>>(
      RT, OBJT, nullptr, (void*)(relx + OBJD), NOBJ, NOBJ, NOBJ, RELIN,
      (long long)NREL*NOBJ, (long long)OBJD*NOBJ, (long long)NREL*RELIN);

  // --- relational MLP: 320 -> 256 -> 256 -> 256 -> 128 (ReLU everywhere) ----
  gemm_bt_kernel<true,false><<<dim3(MREL/TILE_M, HID/TILE_N, 1), 256, 0, stream>>>(
      relx, rwT1, rb1, (void*)hA, RELIN, RELIN, RELIN, HID, 0, 0, 0);
  gemm_bt_kernel<true,false><<<dim3(MREL/TILE_M, HID/TILE_N, 1), 256, 0, stream>>>(
      hA, rwT2, rb2, (void*)hB, HID, HID, HID, HID, 0, 0, 0);
  gemm_bt_kernel<true,false><<<dim3(MREL/TILE_M, HID/TILE_N, 1), 256, 0, stream>>>(
      hB, rwT3, rb3, (void*)hA, HID, HID, HID, HID, 0, 0, 0);
  gemm_bt_kernel<true,false><<<dim3(MREL/TILE_M, EFFD/TILE_N, 1), 256, 0, stream>>>(
      hA, rwT4, rb4, (void*)effects, HID, HID, HID, EFFD, 0, 0, 0);

  // --- aggregate: effect_receivers = R @ effects (K = 16256) ----------------
  transpose_cast_kernel<bf16><<<dim3(EFFD/32, NREL/32, Bb), tb, 0, stream>>>(
      effects, effT, NREL, EFFD, NREL, (long long)NREL*EFFD, (long long)EFFD*NREL);
  gemm_bt_kernel<false,true><<<dim3(NOBJ/TILE_M, EFFD/TILE_N, Bb), 256, 0, stream>>>(
      Rb, effT, nullptr, (void*)effr, NREL, NREL, NREL, EFFD,
      (long long)NOBJ*NREL, (long long)EFFD*NREL, (long long)NOBJ*EFFD);

  // --- assemble obj_x = [objects | external | effect_receivers | 0-pad] -----
  copy_cast_kernel<<<(unsigned)((MOBJ*OBJD + 255) / 256), 256, 0, stream>>>(
      objects, objx, MOBJ*OBJD, OBJD, OBJD, OBJINP);
  copy_cast_kernel<<<(unsigned)((MOBJ*EXTD + 255) / 256), 256, 0, stream>>>(
      ext_eff, objx + OBJD, MOBJ*EXTD, EXTD, EXTD, OBJINP);
  copy_cast_kernel<<<(unsigned)((MOBJ*EFFD + 255) / 256), 256, 0, stream>>>(
      effr, objx + OBJD + EXTD, MOBJ*EFFD, EFFD, EFFD, OBJINP);

  // --- object MLP: 288(+pad) -> 256 -> 256 -> 128 (last: no ReLU, f32) ------
  gemm_bt_kernel<true,false><<<dim3(MOBJ/TILE_M, HID/TILE_N, 1), 256, 0, stream>>>(
      objx, owT1, ob1, (void*)gA, OBJINP, OBJINP, OBJINP, HID, 0, 0, 0);
  gemm_bt_kernel<true,false><<<dim3(MOBJ/TILE_M, HID/TILE_N, 1), 256, 0, stream>>>(
      gA, owT2, ob2, (void*)gB, HID, HID, HID, HID, 0, 0, 0);
  gemm_bt_kernel<false,true><<<dim3(MOBJ/TILE_M, OUTD/TILE_N, 1), 256, 0, stream>>>(
      gB, owT3, ob3, (void*)out, HID, HID, HID, OUTD, 0, 0, 0);
}
